// TensorProductLinear_47837345743207
// MI455X (gfx1250) — matrix operand order repaired and checked by reading, not compiled, not run on hardware
//
#include <hip/hip_runtime.h>
#include <stdint.h>

typedef __attribute__((ext_vector_type(16))) __bf16 v16bf;
typedef __attribute__((ext_vector_type(8)))  float  v8f;

union BFrag {
    uint4 q[2];
    v16bf b;
};

__device__ inline unsigned short f32_to_bf16(float f) {
    unsigned int u = __float_as_uint(f);
    unsigned int r = u + 0x7fffu + ((u >> 16) & 1u);   // round-to-nearest-even
    return (unsigned short)(r >> 16);
}

// ---------------------------------------------------------------------------
// Setup kernel 1: compute the 15 real-basis Clebsch-Gordan (Wigner-3j) tensors
// on device in fp64, exactly mirroring the reference Python. Runs once per
// launch on a single thread (few thousand flops).
// ---------------------------------------------------------------------------
__device__ double dfact(int n) {
    double r = 1.0;
    for (int i = 2; i <= n; ++i) r *= (double)i;
    return r;
}

__device__ double su2_cg(int j1, int m1, int j2, int m2, int j3, int m3) {
    if (m3 != m1 + m2) return 0.0;
    int vmin = -j1 + j2 + m3;
    if (-j1 + m1 > vmin) vmin = -j1 + m1;
    if (0 > vmin) vmin = 0;
    int vmax = j2 + j3 + m1;
    if (j3 - j1 + j2 < vmax) vmax = j3 - j1 + j2;
    if (j3 + m3 < vmax) vmax = j3 + m3;
    double C = sqrt((2.0 * j3 + 1.0) * dfact(j3 + j1 - j2) * dfact(j3 - j1 + j2) *
                    dfact(j1 + j2 - j3) * dfact(j3 + m3) * dfact(j3 - m3) /
                    (dfact(j1 + j2 + j3 + 1) * dfact(j1 - m1) * dfact(j1 + m1) *
                     dfact(j2 - m2) * dfact(j2 + m2)));
    double S = 0.0;
    for (int v = vmin; v <= vmax; ++v) {
        double sgn = ((v + j2 + m2) & 1) ? -1.0 : 1.0;
        S += sgn * dfact(j2 + j3 + m1 - v) * dfact(j1 - m1 + v) /
             (dfact(v) * dfact(j3 - j1 + j2 - v) * dfact(j3 + m3 - v) *
              dfact(v + j1 - j2 - m3));
    }
    return C * S;
}

__device__ void build_q(int l, double qr[5][5], double qi[5][5]) {
    for (int a = 0; a < 5; ++a)
        for (int b = 0; b < 5; ++b) { qr[a][b] = 0.0; qi[a][b] = 0.0; }
    const double s = 0.70710678118654752440;
    for (int mm = -l; mm < 0; ++mm) {
        qr[l + mm][l - mm] = s;       // q[l+mm, l+|mm|] = 1/sqrt2
        qi[l + mm][l + mm] = -s;      // q[l+mm, l-|mm|] = -i/sqrt2
    }
    qr[l][l] = 1.0;
    for (int mm = 1; mm <= l; ++mm) {
        double sg = (mm & 1) ? -1.0 : 1.0;
        qr[l + mm][l + mm] = sg * s;  // (-1)^mm / sqrt2
        qi[l + mm][l - mm] = sg * s;  // i*(-1)^mm / sqrt2
    }
    double cr, ci;                    // (-i)^l
    switch (l & 3) {
        case 0: cr = 1;  ci = 0;  break;
        case 1: cr = 0;  ci = -1; break;
        case 2: cr = -1; ci = 0;  break;
        default: cr = 0; ci = 1;  break;
    }
    for (int a = 0; a < 5; ++a)
        for (int b = 0; b < 5; ++b) {
            double nr = qr[a][b] * cr - qi[a][b] * ci;
            double ni = qr[a][b] * ci + qi[a][b] * cr;
            qr[a][b] = nr; qi[a][b] = ni;
        }
}

__global__ void w3j_setup_kernel(float* __restrict__ w3j) {
    if (threadIdx.x != 0 || blockIdx.x != 0) return;
    const int pl1[15]  = {0, 0, 0,  1, 1, 1, 1, 1, 1,  2, 2, 2, 2, 2, 2};
    const int pl2[15]  = {0, 1, 2,  0, 1, 1, 1, 2, 2,  0, 1, 1, 2, 2, 2};
    const int pl3[15]  = {0, 1, 2,  1, 0, 1, 2, 1, 2,  2, 1, 2, 0, 1, 2};
    const int poff[15] = {0, 1, 10, 35, 44, 53, 80, 125, 170, 245, 270, 315, 390, 415, 490};
    for (int p = 0; p < 15; ++p) {
        int l1 = pl1[p], l2 = pl2[p], l3 = pl3[p];
        int n1 = 2 * l1 + 1, n2 = 2 * l2 + 1, n3 = 2 * l3 + 1;
        double C[5][5][5];
        for (int a = 0; a < 5; ++a)
            for (int b = 0; b < 5; ++b)
                for (int c = 0; c < 5; ++c) C[a][b][c] = 0.0;
        for (int m1 = -l1; m1 <= l1; ++m1)
            for (int m2 = -l2; m2 <= l2; ++m2) {
                int m3 = m1 + m2;
                if (m3 >= -l3 && m3 <= l3)
                    C[l1 + m1][l2 + m2][l3 + m3] = su2_cg(l1, m1, l2, m2, l3, m3);
            }
        double q1r[5][5], q1i[5][5], q2r[5][5], q2i[5][5], q3r[5][5], q3i[5][5];
        build_q(l1, q1r, q1i);
        build_q(l2, q2r, q2i);
        build_q(l3, q3r, q3i);
        // out[j,lq,m] = Re( sum_{i,k,n} Q1[i][j] Q2[k][lq] conj(Q3[n][m]) C[i][k][n] )
        // (the 1/sqrt(2l3+1) inside _su2_cg_tensor cancels the sqrt(2l3+1) applied to W3J)
        for (int j = 0; j < n1; ++j)
            for (int lq = 0; lq < n2; ++lq)
                for (int m = 0; m < n3; ++m) {
                    double sr = 0.0;
                    for (int i = 0; i < n1; ++i)
                        for (int k = 0; k < n2; ++k) {
                            double t1r = q1r[i][j] * q2r[k][lq] - q1i[i][j] * q2i[k][lq];
                            double t1i = q1r[i][j] * q2i[k][lq] + q1i[i][j] * q2r[k][lq];
                            for (int n = 0; n < n3; ++n) {
                                double cv = C[i][k][n];
                                if (cv == 0.0) continue;
                                sr += (t1r * q3r[n][m] + t1i * q3i[n][m]) * cv;
                            }
                        }
                    w3j[poff[p] + (j * n2 + lq) * n3 + m] = (float)sr;
                }
    }
}

// ---------------------------------------------------------------------------
// Setup kernel 2: transpose W (U x 128, f32) -> Wt (128 x U, bf16) so each
// lane's WMMA B-fragment is 16 contiguous bf16 (two global_load_b128).
// ---------------------------------------------------------------------------
__global__ __launch_bounds__(256) void convert_w_kernel(const float* __restrict__ W,
                                                        unsigned short* __restrict__ Wt,
                                                        int U) {
    int e = blockIdx.x * 256 + threadIdx.x;
    if (e >= U * 128) return;
    int v = e / U, u = e - v * U;              // consecutive threads -> consecutive u (coalesced stores)
    Wt[v * U + u] = f32_to_bf16(W[u * 128 + v]);
}

// ---------------------------------------------------------------------------
// Main fused kernel (one instantiation per output l3).
// Block = 256 threads = 8 wave32. M-tile = 16 batch rows; wave w owns v-tile
// [16w, 16w+16). K loop runs over NP paths x 128 mul, 32 at a time via
// v_wmma_f32_16x16x32_bf16 with f32 accumulators resident in VGPRs.
// ---------------------------------------------------------------------------
#define CATP 136   // cat row stride in bf16 elements: 272B = odd multiple of 16B -> conflict-free ds_load_b128

template <int L3>
__global__ __launch_bounds__(256) void tp_linear_wmma(
    const float* __restrict__ x, const float* __restrict__ y,
    const float* __restrict__ w3j, const unsigned short* __restrict__ wt,
    float* __restrict__ out, int Bn) {
    constexpr int K3 = 2 * L3 + 1;
    constexpr int NP = (L3 == 0) ? 3 : 6;
    constexpr int U  = NP * 128;
    constexpr int PL1[3][6]  = {{0, 1, 2, 0, 0, 0}, {0, 1, 1, 1, 2, 2}, {0, 1, 1, 2, 2, 2}};
    constexpr int PL2[3][6]  = {{0, 1, 2, 0, 0, 0}, {1, 0, 1, 2, 1, 2}, {2, 1, 2, 0, 1, 2}};
    constexpr int WOFF[3][6] = {{0, 44, 390, 0, 0, 0},
                                {1, 35, 53, 125, 270, 415},
                                {10, 80, 170, 245, 315, 490}};
    constexpr int XOFF[3] = {0, 128, 512};
    constexpr int YOFF[3] = {0, 1, 4};
    constexpr int OOFF[3] = {0, 128, 512};

    __shared__ float s_y[16][12];
    __shared__ float s_Y[16][28];
    __shared__ __align__(16) unsigned short s_cat[K3 * 16 * CATP];

    const int t    = threadIdx.x;
    const int lane = t & 31;
    const int wv   = t >> 5;
    const int b0   = blockIdx.x * 16;
    const int col  = lane & 15;
    const int hi   = lane >> 4;   // 0: lanes 0-15, 1: lanes 16-31

    if (t < 144) {
        int b = t / 9, j = t - b * 9;
        int bg = b0 + b;
        s_y[b][j] = (bg < Bn) ? y[bg * 9 + j] : 0.0f;
    }

    v8f acc[K3] = {};

#pragma unroll
    for (int pi = 0; pi < NP; ++pi) {
        const int l1 = PL1[L3][pi], l2 = PL2[L3][pi];
        const int n1 = 2 * l1 + 1, n2 = 2 * l2 + 1;
        const float* wj = w3j + WOFF[L3][pi];

        __syncthreads();  // protect s_Y / s_cat from previous iteration readers
        // Y[b][i*K3+k] = sum_j y[b][yoff_l2 + j] * w3j[i,j,k]
        for (int idx = t; idx < 16 * n1 * K3; idx += 256) {
            int b = idx / (n1 * K3);
            int r = idx - b * (n1 * K3);
            int i = r / K3;
            int k = r - i * K3;
            float s = 0.0f;
            for (int j = 0; j < n2; ++j)
                s += s_y[b][YOFF[l2] + j] * wj[(i * n2 + j) * K3 + k];
            s_Y[b][i * K3 + k] = s;
        }
        __syncthreads();
        // cat[k][b][u] = bf16( sum_i x[b,u,i] * Y[b][i*K3+k] )
        for (int idx = t; idx < 2048; idx += 256) {
            int b = idx >> 7, u = idx & 127;
            int bg = b0 + b;
            const float* xp = x + (size_t)bg * 1152 + XOFF[l1] + u * n1;
            float xv[5];
#pragma unroll
            for (int i = 0; i < 5; ++i) xv[i] = (i < n1 && bg < Bn) ? xp[i] : 0.0f;
#pragma unroll
            for (int k = 0; k < K3; ++k) {
                float s = 0.0f;
#pragma unroll
                for (int i = 0; i < 5; ++i)
                    if (i < n1) s += xv[i] * s_Y[b][i * K3 + k];
                s_cat[(k * 16 + b) * CATP + u] = f32_to_bf16(s);
            }
        }
        __syncthreads();
        // WMMA over this path's 128-wide K block, 32 at a time.
#pragma unroll
        for (int ks = 0; ks < 4; ++ks) {
            // B fragment: W[u, v]; lanes 0-15 hold K=0..15, lanes 16-31 K=16..31,
            // two consecutive K packed per VGPR -> 16 contiguous bf16 per lane.
            BFrag bf;
            const unsigned short* wp =
                wt + (size_t)(wv * 16 + col) * U + pi * 128 + ks * 32 + hi * 16;
            bf.q[0] = *reinterpret_cast<const uint4*>(wp);
            bf.q[1] = *reinterpret_cast<const uint4*>(wp + 8);
#pragma unroll
            for (int k = 0; k < K3; ++k) {
                // A fragment: cat[b=col][u]; lanes 0-15: K 0-15, lanes 16-31: K 16-31.
                BFrag af;
                const unsigned short* cp = &s_cat[(k * 16 + col) * CATP + ks * 32 + hi * 16];
                af.q[0] = *reinterpret_cast<const uint4*>(cp);
                af.q[1] = *reinterpret_cast<const uint4*>(cp + 8);
                acc[k] = __builtin_amdgcn_wmma_f32_16x16x32_bf16(
                    false, af.b, false, bf.b, (short)0, acc[k], false, false);
            }
        }
    }

    // Epilogue: D layout -> VGPR r: lanes 0-15 (M=r, N=lane), lanes 16-31 (M=r+8).
    const float scale = (L3 == 0) ? 0.05103103630798288f    // 1/sqrt(384)
                                  : 0.03608439182435161f;   // 1/sqrt(768)
    const int vcol  = wv * 16 + col;
    const int rbase = b0 + (hi ? 8 : 0);
#pragma unroll
    for (int k = 0; k < K3; ++k) {
#pragma unroll
        for (int r = 0; r < 8; ++r) {
            int bg = rbase + r;
            if (bg < Bn)
                out[(size_t)bg * 1152 + OOFF[L3] + vcol * K3 + k] = acc[k][r] * scale;
        }
    }
}

// ---------------------------------------------------------------------------
// Launch: ws layout = [w3j f32: 4KB][Wt0 bf16 384*128][Wt1 bf16 768*128][Wt2 bf16 768*128]
// Total ws use: 4096 + 491520 = 495616 bytes.
// ---------------------------------------------------------------------------
extern "C" void kernel_launch(void* const* d_in, const int* in_sizes, int n_in,
                              void* d_out, int out_size, void* d_ws, size_t ws_size,
                              hipStream_t stream) {
    const float* x  = (const float*)d_in[0];
    const float* y  = (const float*)d_in[1];
    const float* W0 = (const float*)d_in[2];
    const float* W1 = (const float*)d_in[3];
    const float* W2 = (const float*)d_in[4];
    float* out = (float*)d_out;
    const int Bn = in_sizes[0] / 1152;

    float* w3j = (float*)d_ws;
    unsigned short* wt0 = (unsigned short*)((char*)d_ws + 4096);
    unsigned short* wt1 = wt0 + 384 * 128;
    unsigned short* wt2 = wt1 + 768 * 128;

    w3j_setup_kernel<<<1, 32, 0, stream>>>(w3j);
    convert_w_kernel<<<(384 * 128 + 255) / 256, 256, 0, stream>>>(W0, wt0, 384);
    convert_w_kernel<<<(768 * 128 + 255) / 256, 256, 0, stream>>>(W1, wt1, 768);
    convert_w_kernel<<<(768 * 128 + 255) / 256, 256, 0, stream>>>(W2, wt2, 768);

    const int gb = (Bn + 15) / 16;
    tp_linear_wmma<0><<<gb, 256, 0, stream>>>(x, y, w3j, wt0, out, Bn);
    tp_linear_wmma<1><<<gb, 256, 0, stream>>>(x, y, w3j, wt1, out, Bn);
    tp_linear_wmma<2><<<gb, 256, 0, stream>>>(x, y, w3j, wt2, out, Bn);
}
